// Decoder_73916387164678
// MI455X (gfx1250) — compile-verified, matrix operand order checked by reading, not run on hardware
//
#include <hip/hip_runtime.h>
#include <hip/hip_bf16.h>
#include <stdint.h>

// Problem dims
#define D_   512
#define V_   32000
#define B_   32
#define LX_  128
#define TY_  64

typedef __attribute__((ext_vector_type(16))) __bf16 v16bf;
typedef __attribute__((ext_vector_type(8)))  __bf16 v8bf;
typedef __attribute__((ext_vector_type(8)))  float  v8f;

__device__ __forceinline__ float sigm(float x) { return 1.0f / (1.0f + __expf(-x)); }

__device__ __forceinline__ v16bf load_frag_bf16(const __bf16* p, int kb, int klo) {
    // 16-bit WMMA operand per-lane layout (wave32):
    //   lane<16  : elems 0..7 = K{kb..kb+7},   elems 8..15 = K{kb+16..kb+23}
    //   lane>=16 : elems 0..7 = K{kb+8..+15},  elems 8..15 = K{kb+24..+31}
    v8bf lo = *(const v8bf*)(p + kb + klo);
    v8bf hi = *(const v8bf*)(p + kb + 16 + klo);
    return __builtin_shufflevector(lo, hi,
             0, 1, 2, 3, 4, 5, 6, 7, 8, 9, 10, 11, 12, 13, 14, 15);
}

// ---------------------------------------------------------------------------
// Generic f32 -> bf16 conversion (grid-stride)
// ---------------------------------------------------------------------------
__global__ void k_cvt_bf16(const float* __restrict__ s, __bf16* __restrict__ d, long n) {
    long i  = (long)blockIdx.x * blockDim.x + threadIdx.x;
    long st = (long)gridDim.x * blockDim.x;
    for (; i < n; i += st) d[i] = (__bf16)s[i];
}

// emb_bf[m, d] = bf16(word_emb[y_train[m], d]),  m = b*TY + t
__global__ void k_gather_emb(const float* __restrict__ word_emb,
                             const int* __restrict__ y_train,
                             __bf16* __restrict__ dst) {
    int idx = blockIdx.x * blockDim.x + threadIdx.x;       // [0, 2048*512)
    if (idx >= B_ * TY_ * D_) return;
    int m = idx >> 9, d = idx & 511;
    int tok = y_train[m];
    dst[idx] = (__bf16)word_emb[(size_t)tok * D_ + d];
}

// Wg_emb[j, k] = bf16(w_ih[j, k]), k < 512
__global__ void k_build_wge(const float* __restrict__ w_ih, __bf16* __restrict__ dst) {
    int idx = blockIdx.x * blockDim.x + threadIdx.x;       // [0, 2048*512)
    if (idx >= 4 * D_ * D_) return;
    int j = idx >> 9, k = idx & 511;
    dst[idx] = (__bf16)w_ih[(size_t)j * (2 * D_) + k];
}

// Wg_rec[j, k] = bf16( k<512 ? w_ih[j, 512+k] : w_hh[j, k-512] ),  K = 1024
__global__ void k_build_wgr(const float* __restrict__ w_ih, const float* __restrict__ w_hh,
                            __bf16* __restrict__ dst) {
    int idx = blockIdx.x * blockDim.x + threadIdx.x;       // [0, 2048*1024)
    if (idx >= 4 * D_ * 2 * D_) return;
    int j = idx >> 10, k = idx & 1023;
    float v = (k < D_) ? w_ih[(size_t)j * (2 * D_) + D_ + k]
                       : w_hh[(size_t)j * D_ + (k - D_)];
    dst[idx] = (__bf16)v;
}

// Wc1[n, k] = bf16(ctx_to_readout_w[n, k]), k < 512
__global__ void k_build_wc1(const float* __restrict__ ctxw, __bf16* __restrict__ dst) {
    int idx = blockIdx.x * blockDim.x + threadIdx.x;       // [0, 512*512)
    if (idx >= D_ * D_) return;
    int n = idx >> 9, k = idx & 511;
    dst[idx] = (__bf16)ctxw[(size_t)n * (3 * D_) + k];
}

// ---------------------------------------------------------------------------
// WMMA bf16 GEMM, 16(M) x 64(N) tile per wave (for skinny M=32 GEMMs):
//   C[M,N] = A[M,K] * B[N,K]^T   (both row-major, K contiguous)
// ---------------------------------------------------------------------------
__global__ void k_wmma_gemm16(const __bf16* __restrict__ A, const __bf16* __restrict__ Bw,
                              float* __restrict__ C, int M, int N, int K) {
    const int lane   = threadIdx.x & 31;
    const int wid    = blockIdx.x * (blockDim.x >> 5) + (threadIdx.x >> 5);
    const int tilesM = M >> 4;
    const int tilesN = N >> 6;
    if (wid >= tilesM * tilesN) return;
    const int tm = wid % tilesM;
    const int tn = wid / tilesM;

    const int r   = lane & 15;
    const int klo = (lane < 16) ? 0 : 8;

    const __bf16* arow = A  + (size_t)(tm * 16 + r) * K;
    const __bf16* brow = Bw + (size_t)(tn * 64 + r) * K;

    v8f acc[4];
#pragma unroll
    for (int j = 0; j < 4; ++j) acc[j] = v8f{};

    for (int kb = 0; kb < K; kb += 32) {
        v16bf af = load_frag_bf16(arow, kb, klo);
#pragma unroll
        for (int j = 0; j < 4; ++j) {
            v16bf bfr = load_frag_bf16(brow + (size_t)(j * 16) * K, kb, klo);
            acc[j] = __builtin_amdgcn_wmma_f32_16x16x32_bf16(
                         false, af, false, bfr, (short)0, acc[j], false, false);
        }
    }

    // C/D layout: lane<16 -> rows M=0..7 in VGPR 0..7; lane>=16 -> rows M=8..15
    const int moff = (lane < 16) ? 0 : 8;
#pragma unroll
    for (int j = 0; j < 4; ++j) {
        int n0 = tn * 64 + j * 16 + (lane & 15);
#pragma unroll
        for (int rr = 0; rr < 8; ++rr) {
            int m = tm * 16 + moff + rr;
            C[(size_t)m * N + n0] = acc[j][rr];
        }
    }
}

// ---------------------------------------------------------------------------
// WMMA bf16 GEMM, 64(M) x 64(N) tile per wave (for the large GEMMs).
// 4 A-frags x 4 B-frags -> 16 WMMAs per 32-K chunk; cuts L2 re-reads of the
// 33 MB readout matrix by 4x vs the 16x64 tile.
// ---------------------------------------------------------------------------
__global__ void k_wmma_gemm64(const __bf16* __restrict__ A, const __bf16* __restrict__ Bw,
                              float* __restrict__ C, int M, int N, int K) {
    const int lane   = threadIdx.x & 31;
    const int wid    = blockIdx.x * (blockDim.x >> 5) + (threadIdx.x >> 5);
    const int tilesM = M >> 6;
    const int tilesN = N >> 6;
    if (wid >= tilesM * tilesN) return;
    const int tm = wid % tilesM;   // consecutive waves share tn -> B strip L2 locality
    const int tn = wid / tilesM;

    const int r   = lane & 15;
    const int klo = (lane < 16) ? 0 : 8;

    const __bf16* arow = A  + (size_t)(tm * 64 + r) * K;
    const __bf16* brow = Bw + (size_t)(tn * 64 + r) * K;

    v8f acc[16];
#pragma unroll
    for (int i = 0; i < 16; ++i) acc[i] = v8f{};

    for (int kb = 0; kb < K; kb += 32) {
        v16bf af[4], bfr[4];
#pragma unroll
        for (int i = 0; i < 4; ++i)
            af[i] = load_frag_bf16(arow + (size_t)(i * 16) * K, kb, klo);
#pragma unroll
        for (int j = 0; j < 4; ++j)
            bfr[j] = load_frag_bf16(brow + (size_t)(j * 16) * K, kb, klo);
#pragma unroll
        for (int i = 0; i < 4; ++i)
#pragma unroll
            for (int j = 0; j < 4; ++j)
                acc[i * 4 + j] = __builtin_amdgcn_wmma_f32_16x16x32_bf16(
                                     false, af[i], false, bfr[j], (short)0,
                                     acc[i * 4 + j], false, false);
    }

    const int moff = (lane < 16) ? 0 : 8;
#pragma unroll
    for (int i = 0; i < 4; ++i) {
#pragma unroll
        for (int j = 0; j < 4; ++j) {
            int n0 = tn * 64 + j * 16 + (lane & 15);
#pragma unroll
            for (int rr = 0; rr < 8; ++rr) {
                int m = tm * 64 + i * 16 + moff + rr;
                C[(size_t)m * N + n0] = acc[i * 4 + j][rr];
            }
        }
    }
}

// ---------------------------------------------------------------------------
// Attention precompute (t-invariant: softmax is shift-invariant, and the query
// term contributes a per-row constant to the additive scores).
// One block per batch row b. Computes ctx[b, 0:1024].
// ---------------------------------------------------------------------------
__global__ void k_attn(const float* __restrict__ xk, const float* __restrict__ xenc,
                       const unsigned char* __restrict__ xmask,
                       const float* __restrict__ watt, const float* __restrict__ wattb,
                       float* __restrict__ ctx) {
    int b = blockIdx.x, tid = threadIdx.x;
    __shared__ float s_att[LX_];
    if (tid < LX_) {
        const float* row = xk + (size_t)(b * LX_ + tid) * D_;
        float s = 0.f;
        for (int d = 0; d < D_; ++d) s += row[d] * watt[d];
        s += wattb[0];
        if (xmask[b * LX_ + tid]) s = -1e9f;
        s_att[tid] = s;
    }
    __syncthreads();
    if (tid == 0) {
        float m = s_att[0];
        for (int l = 1; l < LX_; ++l) m = fmaxf(m, s_att[l]);
        float sum = 0.f;
        for (int l = 0; l < LX_; ++l) { float e = __expf(s_att[l] - m); s_att[l] = e; sum += e; }
        float inv = 1.f / sum;
        for (int l = 0; l < LX_; ++l) s_att[l] *= inv;
    }
    __syncthreads();
    for (int k = tid; k < 2 * D_; k += blockDim.x) {
        float a = 0.f;
        const float* xb = xenc + (size_t)b * LX_ * (2 * D_) + k;
        for (int l = 0; l < LX_; ++l) a += s_att[l] * xb[(size_t)l * (2 * D_)];
        ctx[b * (2 * D_) + k] = a;
    }
}

// pr_ctx[b, n] = sum_k ctx[b,k] * ctx_to_readout_w[n, 512+k]   (t-invariant)
__global__ void k_prctx(const float* __restrict__ ctx, const float* __restrict__ ctxw,
                        float* __restrict__ prctx) {
    int b = blockIdx.x, tid = threadIdx.x;
    for (int n = tid; n < D_; n += blockDim.x) {
        const float* w  = ctxw + (size_t)n * (3 * D_) + D_;
        const float* cb = ctx + (size_t)b * (2 * D_);
        float a = 0.f;
        for (int k = 0; k < 2 * D_; ++k) a += cb[k] * w[k];
        prctx[b * D_ + n] = a;
    }
}

// h = dec_h0, c = dec_c0, feed = 0
__global__ void k_init(const float* __restrict__ h0, const float* __restrict__ c0,
                       float* __restrict__ h, float* __restrict__ c, float* __restrict__ feed) {
    int idx = blockIdx.x * blockDim.x + threadIdx.x;
    if (idx >= B_ * D_) return;
    h[idx] = h0[idx]; c[idx] = c0[idx]; feed[idx] = 0.f;
}

// A1[b, k] = bf16( k<512 ? feed[b,k] : h[b,k-512] )   (K = 1024)
__global__ void k_prep_rec(const float* __restrict__ feed, const float* __restrict__ h,
                           __bf16* __restrict__ a1) {
    int idx = blockIdx.x * blockDim.x + threadIdx.x;   // [0, 32*1024)
    if (idx >= B_ * 2 * D_) return;
    int b = idx >> 10, k = idx & 1023;
    float v = (k < D_) ? feed[b * D_ + k] : h[b * D_ + (k - D_)];
    a1[idx] = (__bf16)v;
}

// Pointwise LSTM cell: gates = grec + gemb[b*TY+t] + b_ih + b_hh
__global__ void k_lstm(const float* __restrict__ grec, const float* __restrict__ gemb,
                       const float* __restrict__ bih, const float* __restrict__ bhh,
                       float* __restrict__ c, float* __restrict__ h,
                       __bf16* __restrict__ hbf, int t) {
    int idx = blockIdx.x * blockDim.x + threadIdx.x;   // [0, 32*512)
    if (idx >= B_ * D_) return;
    int b = idx >> 9, d = idx & 511;
    const float* gr = grec + (size_t)b * (4 * D_);
    const float* ge = gemb + (size_t)(b * TY_ + t) * (4 * D_);
    float gi = gr[d]          + ge[d]          + bih[d]          + bhh[d];
    float gf = gr[D_ + d]     + ge[D_ + d]     + bih[D_ + d]     + bhh[D_ + d];
    float gg = gr[2 * D_ + d] + ge[2 * D_ + d] + bih[2 * D_ + d] + bhh[2 * D_ + d];
    float go = gr[3 * D_ + d] + ge[3 * D_ + d] + bih[3 * D_ + d] + bhh[3 * D_ + d];
    float cn = sigm(gf) * c[idx] + sigm(gi) * tanhf(gg);
    float hn = sigm(go) * tanhf(cn);
    c[idx] = cn; h[idx] = hn; hbf[idx] = (__bf16)hn;
}

// pre_readout = tanh(pact + pr_ctx); feed <- pre_readout; PR_bf[b*TY+t] <- bf16
__global__ void k_rprep(const float* __restrict__ pact, const float* __restrict__ prctx,
                        float* __restrict__ feed, __bf16* __restrict__ prbf, int t) {
    int idx = blockIdx.x * blockDim.x + threadIdx.x;   // [0, 32*512)
    if (idx >= B_ * D_) return;
    int b = idx >> 9, d = idx & 511;
    float pr = tanhf(pact[idx] + prctx[idx]);
    feed[idx] = pr;
    prbf[(size_t)(b * TY_ + t) * D_ + d] = (__bf16)pr;
}

// ---------------------------------------------------------------------------
// Host side
// ---------------------------------------------------------------------------
static inline void launch_gemm(const __bf16* A, const __bf16* B, float* C,
                               int M, int N, int K, hipStream_t stream) {
    if ((M & 63) == 0 && (N & 63) == 0) {
        int waves  = (M >> 6) * (N >> 6);
        int blocks = (waves + 3) / 4;
        hipLaunchKernelGGL(k_wmma_gemm64, dim3(blocks), dim3(128), 0, stream, A, B, C, M, N, K);
    } else {
        int waves  = (M >> 4) * (N >> 6);
        int blocks = (waves + 3) / 4;
        hipLaunchKernelGGL(k_wmma_gemm16, dim3(blocks), dim3(128), 0, stream, A, B, C, M, N, K);
    }
}

extern "C" void kernel_launch(void* const* d_in, const int* in_sizes, int n_in,
                              void* d_out, int out_size, void* d_ws, size_t ws_size,
                              hipStream_t stream) {
    (void)in_sizes; (void)n_in; (void)out_size; (void)ws_size;

    const float* x_enc  = (const float*)d_in[0];
    const float* dec_h0 = (const float*)d_in[1];
    const float* dec_c0 = (const float*)d_in[2];
    const unsigned char* x_mask = (const unsigned char*)d_in[3];   // all-false bools
    const int*   y_train = (const int*)d_in[4];
    // d_in[5] = y_mask (unused by the reference)
    const float* enc_to_k_w = (const float*)d_in[6];
    // w_trg_w / w_trg_b (d_in[7], d_in[8]) only shift attention scores by a
    // per-row constant -> drop out of softmax; unused.
    const float* w_att_w = (const float*)d_in[9];
    const float* w_att_b = (const float*)d_in[10];
    const float* ctx_to_readout_w = (const float*)d_in[11];
    const float* readout_w = (const float*)d_in[12];
    const float* word_emb  = (const float*)d_in[13];
    const float* w_ih = (const float*)d_in[14];
    const float* w_hh = (const float*)d_in[15];
    const float* b_ih = (const float*)d_in[16];
    const float* b_hh = (const float*)d_in[17];
    float* out = (float*)d_out;

    // ---- workspace partition (256 B aligned slices) ----
    char* cur = (char*)d_ws;
    auto take = [&](size_t bytes) -> void* {
        uintptr_t p = (uintptr_t)cur;
        p = (p + 255) & ~(uintptr_t)255;
        cur = (char*)(p + bytes);
        return (void*)p;
    };
    __bf16* xenc_bf = (__bf16*)take((size_t)B_ * LX_ * 2 * D_ * 2);   // 8.4 MB
    __bf16* ekw_bf  = (__bf16*)take((size_t)D_ * 2 * D_ * 2);         // 1 MB
    float*  xk      = (float*) take((size_t)B_ * LX_ * D_ * 4);       // 8.4 MB
    __bf16* emb_bf  = (__bf16*)take((size_t)B_ * TY_ * D_ * 2);       // 2.1 MB
    __bf16* wge     = (__bf16*)take((size_t)4 * D_ * D_ * 2);         // 2.1 MB
    __bf16* wgr     = (__bf16*)take((size_t)4 * D_ * 2 * D_ * 2);     // 4.2 MB
    __bf16* wc1     = (__bf16*)take((size_t)D_ * D_ * 2);             // 0.5 MB
    __bf16* rdw     = (__bf16*)take((size_t)V_ * D_ * 2);             // 32.8 MB
    float*  gemb    = (float*) take((size_t)B_ * TY_ * 4 * D_ * 4);   // 16.8 MB
    float*  ctx     = (float*) take((size_t)B_ * 2 * D_ * 4);
    float*  prctx   = (float*) take((size_t)B_ * D_ * 4);
    float*  h       = (float*) take((size_t)B_ * D_ * 4);
    float*  c       = (float*) take((size_t)B_ * D_ * 4);
    float*  feed    = (float*) take((size_t)B_ * D_ * 4);
    __bf16* a1      = (__bf16*)take((size_t)B_ * 2 * D_ * 2);
    __bf16* hbf     = (__bf16*)take((size_t)B_ * D_ * 2);
    float*  grec    = (float*) take((size_t)B_ * 4 * D_ * 4);
    float*  pact    = (float*) take((size_t)B_ * D_ * 4);
    __bf16* prbf    = (__bf16*)take((size_t)B_ * TY_ * D_ * 2);       // 2.1 MB

    // ---- one-time precompute (stream-ordered) ----
    hipLaunchKernelGGL(k_cvt_bf16, dim3(4096), dim3(256), 0, stream,
                       x_enc, xenc_bf, (long)B_ * LX_ * 2 * D_);
    hipLaunchKernelGGL(k_cvt_bf16, dim3(1024), dim3(256), 0, stream,
                       enc_to_k_w, ekw_bf, (long)D_ * 2 * D_);
    hipLaunchKernelGGL(k_cvt_bf16, dim3(8192), dim3(256), 0, stream,
                       readout_w, rdw, (long)V_ * D_);
    hipLaunchKernelGGL(k_gather_emb, dim3((B_ * TY_ * D_) / 256), dim3(256), 0, stream,
                       word_emb, y_train, emb_bf);
    hipLaunchKernelGGL(k_build_wge, dim3((4 * D_ * D_) / 256), dim3(256), 0, stream,
                       w_ih, wge);
    hipLaunchKernelGGL(k_build_wgr, dim3((4 * D_ * 2 * D_) / 256), dim3(256), 0, stream,
                       w_ih, w_hh, wgr);
    hipLaunchKernelGGL(k_build_wc1, dim3((D_ * D_) / 256), dim3(256), 0, stream,
                       ctx_to_readout_w, wc1);

    // x_enc_k = x_enc @ enc_to_k_w^T     [4096,512] , K=1024   (64x64 kernel)
    launch_gemm(xenc_bf, ekw_bf, xk, B_ * LX_, D_, 2 * D_, stream);
    // gates_emb = emb @ w_ih[:, :512]^T  [2048,2048], K=512    (all steps at once)
    launch_gemm(emb_bf, wge, gemb, B_ * TY_, 4 * D_, D_, stream);

    hipLaunchKernelGGL(k_attn, dim3(B_), dim3(256), 0, stream,
                       xk, x_enc, x_mask, w_att_w, w_att_b, ctx);
    hipLaunchKernelGGL(k_prctx, dim3(B_), dim3(256), 0, stream,
                       ctx, ctx_to_readout_w, prctx);
    hipLaunchKernelGGL(k_init, dim3((B_ * D_) / 256), dim3(256), 0, stream,
                       dec_h0, dec_c0, h, c, feed);

    // ---- sequential recurrence: stream-ordered launches = inter-step sync ----
    for (int t = 0; t < TY_; ++t) {
        hipLaunchKernelGGL(k_prep_rec, dim3((B_ * 2 * D_) / 256), dim3(256), 0, stream,
                           feed, h, a1);
        // gates_rec = [feed|h] @ [w_ih[:,512:]|w_hh]^T  [32,2048], K=1024
        launch_gemm(a1, wgr, grec, B_, 4 * D_, 2 * D_, stream);
        hipLaunchKernelGGL(k_lstm, dim3((B_ * D_) / 256), dim3(256), 0, stream,
                           grec, gemb, b_ih, b_hh, c, h, hbf, t);
        // pre_act = h @ ctx_to_readout_w[:, :512]^T     [32,512], K=512
        launch_gemm(hbf, wc1, pact, B_, D_, D_, stream);
        hipLaunchKernelGGL(k_rprep, dim3((B_ * D_) / 256), dim3(256), 0, stream,
                           pact, prctx, feed, prbf, t);
    }

    // ---- batched readout: logits[b*TY+t, v] = PR @ readout_w^T  [2048,32000], K=512
    launch_gemm(prbf, rdw, out, B_ * TY_, V_, D_, stream);
}
